// ConvMod1D_27513560498550
// MI455X (gfx1250) — matrix operand order repaired and checked by reading, not compiled, not run on hardware
//
#include <hip/hip_runtime.h>
#include <hip/hip_bf16.h>
#include <math.h>

// ---------------------------------------------------------------------------
// ConvMod1D for MI455X (gfx1250, wave32).
// Pointwise convs = per-batch GEMMs W(OxC) @ X(CxL) via
// V_WMMA_F32_16X16X32_BF16 (fp32 -> bf16, fp32 accumulate).
//  - weights staged block-wide in LDS as bf16 (ds_load_b128 A-fragments)
//  - each wave owns 4 O-tiles: one B-fragment feeds 4 WMMAs per K-step
// LayerNorm / depthwise / gating are scalar fp32 (bandwidth-bound).
// ---------------------------------------------------------------------------

typedef __attribute__((ext_vector_type(16))) __bf16 v16bf;
typedef __attribute__((ext_vector_type(8)))  __bf16 v8bf;
typedef __attribute__((ext_vector_type(8)))  float  v8f;

constexpr int       Bc  = 32;
constexpr int       Dc  = 256;
constexpr int       Lc  = 4096;
constexpr int       Qc  = Dc / 4;      // 64
constexpr int       Hc  = Dc / 2;      // 128
constexpr int       Tc  = 3 * Dc / 4;  // 192
constexpr long long BLL = (long long)Bc * Lc;  // 131072

// ---------------------------------------------------------------------------
// Pointwise conv as WMMA GEMM:  Y[b] = W(OxC) @ X[b](CxL) + bias (opt. GELU)
// Block = 128 threads (4 waves).  Block owns a 64(O) x 64(L) output tile:
// wave w takes L-subtile l0 = bx*64 + 16w and all four 16-row O-subtiles.
// grid = (L/64, O/64, B);  dynamic LDS = 64*C bf16 (weight slab).
// ---------------------------------------------------------------------------
__global__ void __launch_bounds__(128)
pw_wmma_kernel(const float* __restrict__ X, long long x_bstride, int C,
               const float* __restrict__ W, const float* __restrict__ bias,
               float* __restrict__ Y, long long y_bstride, int do_gelu)
{
    extern __shared__ __bf16 ldsW[];  // [64][C] row-major bf16

    const int lane   = threadIdx.x & 31;
    const int wave   = threadIdx.x >> 5;
    const int l0     = blockIdx.x * 64 + wave * 16;
    const int o_base = blockIdx.y * 64;
    const int b      = blockIdx.z;

    // ---- cooperative stage: W[o_base:o_base+64][0:C] -> LDS (fp32->bf16) ----
    for (int i = threadIdx.x; i < 64 * C; i += 128) {
        const int row = i / C;
        const int col = i - row * C;
        ldsW[i] = (__bf16)W[(long long)(o_base + row) * C + col];
    }
    __syncthreads();

    const int m = lane & 15;   // B/D column (and A row) within tile
    const int h = lane >> 4;   // lane half-select

    const float* Xb = X + (long long)b * x_bstride;

    v8f acc0 = {}, acc1 = {}, acc2 = {}, acc3 = {};

    for (int k0 = 0; k0 < C; k0 += 32) {
        // ---- B fragment: X[k0+kk][l0+m], contiguous k = 16h + s (same as A) ----
        const float* xcol = Xb + (long long)k0 * Lc + l0 + m;
        v16bf bfrag;
#pragma unroll
        for (int r = 0; r < 8; ++r) {
            const int kk = (h << 4) + (r << 1);
            bfrag[2 * r]     = (__bf16)xcol[(long long)kk * Lc];
            bfrag[2 * r + 1] = (__bf16)xcol[(long long)(kk + 1) * Lc];
        }

        if (k0 + 32 < C) {
            // gfx1250 global_prefetch_b8 of the next K-slab of X
            __builtin_prefetch(xcol + 32LL * Lc, 0, 0);
        }

        // ---- 4 A fragments from LDS; reuse bfrag across 4 WMMAs ----
#pragma unroll
        for (int t = 0; t < 4; ++t) {
            const int row = t * 16 + m;
            // A layout: afrag[0..7]  = W[row][k0 + 16h + 0..7]
            //           afrag[8..15] = W[row][k0 + 16h + 8 + 0..7]
            const __bf16* wp = ldsW + row * C + k0 + (h << 4);
            const v8bf lo = *(const v8bf*)(wp);
            const v8bf hi = *(const v8bf*)(wp + 8);
            v16bf afrag;
#pragma unroll
            for (int i = 0; i < 8; ++i) {
                afrag[i]     = lo[i];
                afrag[8 + i] = hi[i];
            }
            v8f* accp = (t == 0) ? &acc0 : (t == 1) ? &acc1 : (t == 2) ? &acc2 : &acc3;
            *accp = __builtin_amdgcn_wmma_f32_16x16x32_bf16(
                false, afrag, false, bfrag, (short)0, *accp, false, false);
        }
    }

    // ---- Epilogue: bias (+ exact GELU), store per C/D 16x16 f32 layout ----
    const int n = m;  // D column
#pragma unroll
    for (int t = 0; t < 4; ++t) {
        const v8f acc = (t == 0) ? acc0 : (t == 1) ? acc1 : (t == 2) ? acc2 : acc3;
#pragma unroll
        for (int r = 0; r < 8; ++r) {
            const int row = o_base + t * 16 + r + (h << 3);
            float v = acc[r] + bias[row];
            if (do_gelu)
                v = 0.5f * v * (1.0f + erff(v * 0.70710678118654752f));
            Y[(long long)b * y_bstride + (long long)row * Lc + l0 + n] = v;
        }
    }
}

// ---------------------------------------------------------------------------
// Channel-first LayerNorm over C channels. One thread per (b, l).
// ---------------------------------------------------------------------------
__global__ void __launch_bounds__(256)
ln_cf_kernel(const float* __restrict__ X, const float* __restrict__ w,
             const float* __restrict__ bvec, float* __restrict__ Y, int C)
{
    const long long gid = (long long)blockIdx.x * blockDim.x + threadIdx.x;
    if (gid >= BLL) return;
    const int b = (int)(gid / Lc);
    const int l = (int)(gid % Lc);

    const float* xb = X + (long long)b * C * Lc + l;
    float s = 0.f, s2 = 0.f;
    for (int c = 0; c < C; ++c) {
        const float v = xb[(long long)c * Lc];
        s  += v;
        s2 += v * v;
    }
    const float mean = s / (float)C;
    const float var  = s2 / (float)C - mean * mean;
    const float inv  = rsqrtf(var + 1e-6f);

    float* yb = Y + (long long)b * C * Lc + l;
    for (int c = 0; c < C; ++c) {
        const float v = xb[(long long)c * Lc];
        yb[(long long)c * Lc] = w[c] * (v - mean) * inv + bvec[c];
    }
}

// ---------------------------------------------------------------------------
// Depthwise conv1d ("same" padding) + bias + optional residual add.
// ---------------------------------------------------------------------------
__global__ void __launch_bounds__(256)
dw_conv_kernel(const float* __restrict__ Xin, long long x_bstride, int C,
               const float* __restrict__ w, const float* __restrict__ bias,
               int K, int pad,
               const float* __restrict__ add, long long add_bstride,
               float* __restrict__ Y, long long y_bstride)
{
    const long long total = (long long)Bc * C * Lc;
    const long long gid   = (long long)blockIdx.x * blockDim.x + threadIdx.x;
    if (gid >= total) return;
    const int l = (int)(gid % Lc);
    const int c = (int)((gid / Lc) % C);
    const int b = (int)(gid / ((long long)C * Lc));

    const float* xc = Xin + (long long)b * x_bstride + (long long)c * Lc;
    float acc = bias[c];
    for (int j = 0; j < K; ++j) {
        const int pos = l + j - pad;
        if (pos >= 0 && pos < Lc) acc += w[c * K + j] * xc[pos];
    }
    if (add) acc += add[(long long)b * add_bstride + (long long)c * Lc + l];
    Y[(long long)b * y_bstride + (long long)c * Lc + l] = acc;
}

// ---------------------------------------------------------------------------
// Elementwise gating:  dst *= src
// ---------------------------------------------------------------------------
__global__ void __launch_bounds__(256)
mul_kernel(float* __restrict__ dst, const float* __restrict__ src, long long n)
{
    const long long gid = (long long)blockIdx.x * blockDim.x + threadIdx.x;
    if (gid < n) dst[gid] *= src[gid];
}

// ---------------------------------------------------------------------------
// Host-side pipeline
// ---------------------------------------------------------------------------
static inline void launch_pw(const float* X, long long xbs, int C,
                             const float* W, const float* bias,
                             float* Y, long long ybs, int O, int gelu,
                             hipStream_t s)
{
    dim3 grid(Lc / 64, O / 64, Bc);
    const size_t shmem = (size_t)64 * C * sizeof(__bf16);
    pw_wmma_kernel<<<grid, 128, shmem, s>>>(X, xbs, C, W, bias, Y, ybs, gelu);
}

static inline void launch_dw(const float* X, long long xbs, int C,
                             const float* w, const float* bias, int K, int pad,
                             const float* add, long long abs_,
                             float* Y, long long ybs, hipStream_t s)
{
    const long long total = (long long)Bc * C * Lc;
    dw_conv_kernel<<<(unsigned)((total + 255) / 256), 256, 0, s>>>(
        X, xbs, C, w, bias, K, pad, add, abs_, Y, ybs);
}

extern "C" void kernel_launch(void* const* d_in, const int* in_sizes, int n_in,
                              void* d_out, int out_size, void* d_ws, size_t ws_size,
                              hipStream_t stream)
{
    const float* x = (const float*)d_in[0];
#define IN(i) ((const float*)d_in[(i)])
    const float *ln1_w = IN(1),  *ln1_b = IN(2);
    const float *a1p_w = IN(3),  *a1p_b = IN(4);
    const float *a1d_w = IN(5),  *a1d_b = IN(6);
    const float *v1_w  = IN(7),  *v1_b  = IN(8);
    const float *v11_w = IN(9),  *v11_b = IN(10);
    const float *v12_w = IN(11), *v12_b = IN(12);
    const float *c31_w = IN(13), *c31_b = IN(14);
    const float *ln2_w = IN(15), *ln2_b = IN(16);
    const float *a2p_w = IN(17), *a2p_b = IN(18);
    const float *a2d_w = IN(19), *a2d_b = IN(20);
    const float *v2_w  = IN(21), *v2_b  = IN(22);
    const float *v21_w = IN(23), *v21_b = IN(24);
    const float *v22_w = IN(25), *v22_b = IN(26);
    const float *p2_w  = IN(27), *p2_b  = IN(28);
    const float *c32_w = IN(29), *c32_b = IN(30);
    const float *ln3_w = IN(31), *ln3_b = IN(32);
    const float *a3p_w = IN(33), *a3p_b = IN(34);
    const float *a3d_w = IN(35), *a3d_b = IN(36);
    const float *v3_w  = IN(37), *v3_b  = IN(38);
    const float *v31_w = IN(39), *v31_b = IN(40);
    const float *v32_w = IN(41), *v32_b = IN(42);
    const float *p3_w  = IN(43), *p3_b  = IN(44);
    const float *c33_w = IN(45), *c33_b = IN(46);
#undef IN

    float* out = (float*)d_out;
    float* ws  = (float*)d_ws;

    // Workspace arena (floats).
    float* XN   = ws;                          // [B, D, L]   LN1(x); xs slices
    float* XIN  = XN   + (long long)Dc * BLL;  // [B, <=T, L] LN2/LN3 output
    float* ABUF = XIN  + (long long)Tc * BLL;  // [B, <=T, L] post-dw "a"
    float* TV   = ABUF + (long long)Tc * BLL;  // [B, <=T, L] gelu-pre / v / prod
    float* TB   = TV   + (long long)Tc * BLL;  // [B, Q, L]   branch temp
    float* PB   = TB   + (long long)Qc * BLL;  // [B, Q, L]   p2/p3 temp
    float* OUT1 = PB   + (long long)Qc * BLL;  // [B, <=T, L] x1 / x2
    const size_t need = (size_t)((long long)(Dc + 4 * Tc + 2 * Qc) * BLL) * sizeof(float);
    if (ws_size < need) return;  // workspace too small: nothing safe to do

    const long long sD = (long long)Dc * Lc, sH = (long long)Hc * Lc,
                    sT = (long long)Tc * Lc, sQ = (long long)Qc * Lc;
    const unsigned lnBlocks = (unsigned)((BLL + 255) / 256);

    // ---------------- LN1 ----------------
    ln_cf_kernel<<<lnBlocks, 256, 0, stream>>>(x, ln1_w, ln1_b, XN, Dc);
    const float* xs0 = XN;
    const float* xs1 = XN + 1 * sQ;
    const float* xs2 = XN + 2 * sQ;
    const float* xs3 = XN + 3 * sQ;

    // ---------------- stage 1 (C = Q) ----------------
    launch_pw(xs0, sD, Qc, a1p_w, a1p_b, TV, sQ, Qc, 1, stream);             // gelu(pw)
    launch_dw(TV, sQ, Qc, a1d_w, a1d_b, 7, 3, nullptr, 0, ABUF, sQ, stream); // a
    launch_pw(xs0, sD, Qc, v1_w, v1_b, TV, sQ, Qc, 0, stream);               // v
    mul_kernel<<<(unsigned)((Qc * BLL + 255) / 256), 256, 0, stream>>>(TV, ABUF, Qc * BLL);
    launch_pw(TV, sQ, Qc, v11_w, v11_b, OUT1 + sQ, sH, Qc, 0, stream);       // mul -> x1[Q:H]
    launch_pw(xs1, sD, Qc, v12_w, v12_b, TB, sQ, Qc, 0, stream);
    launch_dw(TB, sQ, Qc, c31_w, c31_b, 3, 1, ABUF, sQ, OUT1, sH, stream);   // x1[0:Q]

    // ---------------- stage 2 (C = H) ----------------
    ln_cf_kernel<<<lnBlocks, 256, 0, stream>>>(OUT1, ln2_w, ln2_b, XIN, Hc);
    launch_pw(XIN, sH, Hc, a2p_w, a2p_b, TV, sH, Hc, 1, stream);
    launch_dw(TV, sH, Hc, a2d_w, a2d_b, 9, 4, nullptr, 0, ABUF, sH, stream);
    launch_pw(XIN, sH, Hc, v2_w, v2_b, TV, sH, Hc, 0, stream);
    mul_kernel<<<(unsigned)((Hc * BLL + 255) / 256), 256, 0, stream>>>(TV, ABUF, Hc * BLL);
    launch_pw(TV, sH, Hc, v21_w, v21_b, OUT1 + sQ, sT, Hc, 0, stream);       // mul -> x2[Q:T]
    launch_pw(xs2, sD, Qc, v22_w, v22_b, TB, sQ, Qc, 0, stream);
    launch_pw(ABUF, sH, Hc, p2_w, p2_b, PB, sQ, Qc, 0, stream);              // pw(a, p2)
    launch_dw(TB, sQ, Qc, c32_w, c32_b, 3, 1, PB, sQ, OUT1, sT, stream);     // x2[0:Q]

    // ---------------- stage 3 (C = T) ----------------
    ln_cf_kernel<<<lnBlocks, 256, 0, stream>>>(OUT1, ln3_w, ln3_b, XIN, Tc);
    launch_pw(XIN, sT, Tc, a3p_w, a3p_b, TV, sT, Tc, 1, stream);
    launch_dw(TV, sT, Tc, a3d_w, a3d_b, 11, 5, nullptr, 0, ABUF, sT, stream);
    launch_pw(XIN, sT, Tc, v3_w, v3_b, TV, sT, Tc, 0, stream);
    mul_kernel<<<(unsigned)((Tc * BLL + 255) / 256), 256, 0, stream>>>(TV, ABUF, Tc * BLL);
    launch_pw(TV, sT, Tc, v31_w, v31_b, out + sQ, sD, Tc, 0, stream);        // mul -> out[Q:D]
    launch_pw(xs3, sD, Qc, v32_w, v32_b, TB, sQ, Qc, 0, stream);
    launch_pw(ABUF, sT, Tc, p3_w, p3_b, PB, sQ, Qc, 0, stream);              // pw(a, p3)
    launch_dw(TB, sQ, Qc, c33_w, c33_b, 3, 1, PB, sQ, out, sD, stream);      // out[0:Q]
}
